// HierarchicalSparseAttentionTriton_85813446574397
// MI455X (gfx1250) — compile-verified, matrix operand order checked by reading, not run on hardware
//
#include <hip/hip_runtime.h>

// ---------------- problem constants (match reference) ----------------
#define DIMC   1024
#define NB     2
#define NSEQ   4096
#define NH     16
#define DHEAD  64
#define NLEV   12
#define TOTN   (2*NSEQ - 1)   // 8191 nodes per batch
#define SCALEF 0.125f         // 1/sqrt(64)

typedef __bf16 bf16;
typedef __attribute__((ext_vector_type(8)))  bf16  v8bf;
typedef __attribute__((ext_vector_type(16))) bf16  v16bf;
typedef __attribute__((ext_vector_type(8)))  float v8f;

// gfx1250 async global->LDS staging (ASYNCcnt path), with safe fallback
#if defined(__gfx1250__) && __has_builtin(__builtin_amdgcn_global_load_async_to_lds_b128) && \
    __has_builtin(__builtin_amdgcn_s_wait_asynccnt)
#define USE_ASYNC_LDS 1
typedef int v4i __attribute__((vector_size(16)));
typedef __attribute__((address_space(1))) v4i* gbl_v4i_p;
typedef __attribute__((address_space(3))) v4i* lds_v4i_p;
__device__ __forceinline__ void async_ld_b128(const void* g, void* l) {
    __builtin_amdgcn_global_load_async_to_lds_b128(
        (gbl_v4i_p)(void*)g, (lds_v4i_p)l, 0, 0);
}
#else
#define USE_ASYNC_LDS 0
#endif

// ---------------- tiny elementwise kernels ----------------

// Wt[n*1024+k] = (bf16) W[k*1024+n]  (pre-transposed weights for B fragments)
__global__ void k_transpose_w(const float* __restrict__ W, bf16* __restrict__ Wt) {
    int id = blockIdx.x * blockDim.x + threadIdx.x;
    if (id >= DIMC * DIMC) return;
    int n = id >> 10, k = id & (DIMC - 1);
    Wt[id] = (bf16)W[k * DIMC + n];
}

// copy x (fp32, [B,N,DIM]) into the leaf region of h_all (bf16, [B,TOTN,DIM])
__global__ void k_x_to_hall(const float* __restrict__ x, bf16* __restrict__ hall) {
    int id = blockIdx.x * blockDim.x + threadIdx.x;
    if (id >= NB * NSEQ * DIMC) return;
    int b = id / (NSEQ * DIMC);
    int r = id - b * (NSEQ * DIMC);
    hall[(size_t)b * TOTN * DIMC + r] = (bf16)x[id];
}

// pre[b,p,:] = 0.5*(cur[b,2p,:] + cur[b,2p+1,:]); cur batch-strided in h_all
__global__ void k_pair_avg(const bf16* __restrict__ cur, bf16* __restrict__ pre, int P) {
    int id = blockIdx.x * blockDim.x + threadIdx.x;
    int total = NB * P * (DIMC / 8);
    if (id >= total) return;
    int ch  = id & 127;          // chunk of 8 elements
    int row = id >> 7;           // b*P + p
    int b = row / P, p = row - b * P;
    const bf16* s0 = cur + (size_t)b * TOTN * DIMC + (size_t)(2 * p) * DIMC + ch * 8;
    v8bf a = *(const v8bf*)s0;
    v8bf c = *(const v8bf*)(s0 + DIMC);
    v8bf o;
#pragma unroll
    for (int e = 0; e < 8; ++e) o[e] = (bf16)(0.5f * ((float)a[e] + (float)c[e]));
    *(v8bf*)(pre + (size_t)row * DIMC + ch * 8) = o;
}

// ---------------- WMMA GEMM: C[M,1024] = A[M,1024] @ Wt^T (+bias) ----------------
// Tile: 128(M) x 64(N) x 32(K) per 128-thread block (4 waves).
// Wave w owns rows [32w, 32w+32): 2 A fragments x 4 B fragments = 8 WMMA / K-step.
// Out-of-range M rows are CLAMPED (not zeroed): each output row depends only on
// its own A row, and clamped rows are never stored -> unconditional staging loads.
#define LDT 40   // padded LDS row stride (elements); 80B rows keep 16B alignment

template <bool OUT_F32, bool HAS_BIAS>
__global__ __launch_bounds__(128)
void k_gemm_wmma(const bf16* __restrict__ A, long strideA, int rowsPB,
                 const bf16* __restrict__ Wt, const float* __restrict__ bias,
                 void* __restrict__ Cv, long strideC, int M) {
    __shared__ bf16 ldsA[128 * LDT];
    __shared__ bf16 ldsB[64 * LDT];

    const int t    = threadIdx.x;
    const int lane = t & 31;
    const int w    = t >> 5;       // wave 0..3 -> C rows 32w..32w+31
    const int rm   = lane & 15;    // row/col within 16
    const int kh   = lane >> 4;    // K-half select (ISA 16-bit fragment layout)
    const int GM0  = blockIdx.x * 128;
    const int GN0  = blockIdx.y * 64;

    // ---- hoisted per-thread staging addresses (row-clamped, division out of loop)
    // A: 128 rows x 4 chunks of 8 bf16 = 512 chunks; thread t stages chunks t+128*c
    const bf16* aSrc[4];
    int         aDst[4];
#pragma unroll
    for (int c = 0; c < 4; ++c) {
        int cid = t + 128 * c;
        int row = cid >> 2, ch = cid & 3;
        int gm = GM0 + row;
        gm = gm < M ? gm : (M - 1);              // clamp, never stored
        int bb = gm / rowsPB, ii = gm - bb * rowsPB;
        aSrc[c] = A + (size_t)bb * strideA + (size_t)ii * DIMC + ch * 8;
        aDst[c] = row * LDT + ch * 8;
    }
    // B: 64 cols x 4 chunks = 256 chunks; thread t stages chunks t+128*c
    const bf16* bSrc[2];
    int         bDst[2];
#pragma unroll
    for (int c = 0; c < 2; ++c) {
        int cid = t + 128 * c;
        int row = cid >> 2, ch = cid & 3;
        bSrc[c] = Wt + (size_t)(GN0 + row) * DIMC + ch * 8;
        bDst[c] = row * LDT + ch * 8;
    }

    v8f acc[2][4];
#pragma unroll
    for (int s = 0; s < 2; ++s)
#pragma unroll
        for (int j = 0; j < 4; ++j)
#pragma unroll
            for (int e = 0; e < 8; ++e) acc[s][j][e] = 0.0f;

    for (int k0 = 0; k0 < DIMC; k0 += 32) {
#if USE_ASYNC_LDS
#pragma unroll
        for (int c = 0; c < 4; ++c)
            async_ld_b128(aSrc[c] + k0, &ldsA[aDst[c]]);
#pragma unroll
        for (int c = 0; c < 2; ++c)
            async_ld_b128(bSrc[c] + k0, &ldsB[bDst[c]]);
        __builtin_amdgcn_s_wait_asynccnt(0);
#else
#pragma unroll
        for (int c = 0; c < 4; ++c)
            *(v8bf*)&ldsA[aDst[c]] = *(const v8bf*)(aSrc[c] + k0);
#pragma unroll
        for (int c = 0; c < 2; ++c)
            *(v8bf*)&ldsB[bDst[c]] = *(const v8bf*)(bSrc[c] + k0);
#endif
        if (k0 + 32 < DIMC) __builtin_prefetch(bSrc[0] + k0 + 32, 0, 0);  // global_prefetch_b8
        __syncthreads();

        // B fragments shared across this wave's two row-tiles
        v16bf bfrag[4];
#pragma unroll
        for (int j = 0; j < 4; ++j) {
            const bf16* br = &ldsB[(16 * j + rm) * LDT];
            v8bf blo = *(const v8bf*)(br + kh * 8);
            v8bf bhi = *(const v8bf*)(br + 16 + kh * 8);
            bfrag[j] = __builtin_shufflevector(blo, bhi, 0, 1, 2, 3, 4, 5, 6, 7,
                                               8, 9, 10, 11, 12, 13, 14, 15);
        }
#pragma unroll
        for (int s = 0; s < 2; ++s) {
            const bf16* ar = &ldsA[(32 * w + 16 * s + rm) * LDT];
            v8bf alo = *(const v8bf*)(ar + kh * 8);
            v8bf ahi = *(const v8bf*)(ar + 16 + kh * 8);
            v16bf afrag = __builtin_shufflevector(alo, ahi, 0, 1, 2, 3, 4, 5, 6, 7,
                                                  8, 9, 10, 11, 12, 13, 14, 15);
#pragma unroll
            for (int j = 0; j < 4; ++j)
                acc[s][j] = __builtin_amdgcn_wmma_f32_16x16x32_bf16(
                    false, afrag, false, bfrag[j], (short)0, acc[s][j], false, false);
        }
        __syncthreads();
    }

    // store: element (m,n) lives in lane = (n&15)+16*(m>>3), vgpr = m&7
#pragma unroll
    for (int s = 0; s < 2; ++s)
#pragma unroll
        for (int j = 0; j < 4; ++j) {
            int gn = GN0 + 16 * j + rm;
            float bv = HAS_BIAS ? bias[gn] : 0.0f;
#pragma unroll
            for (int e = 0; e < 8; ++e) {
                int ml = (kh << 3) + e;
                int gm = GM0 + 32 * w + 16 * s + ml;
                if (gm < M) {
                    int bb = gm / rowsPB, ii = gm - bb * rowsPB;
                    float val = acc[s][j][e] + bv;
                    size_t o = (size_t)bb * strideC + (size_t)ii * DIMC + gn;
                    if constexpr (OUT_F32) ((float*)Cv)[o] = val;
                    else                   ((bf16*)Cv)[o] = (bf16)val;
                }
            }
        }
}

// ---------------- per-head helpers ----------------

__device__ inline float dot64(const v8bf* __restrict__ q, const bf16* __restrict__ kp) {
    float s = 0.0f;
#pragma unroll
    for (int c = 0; c < 8; ++c) {
        v8bf kv = *(const v8bf*)(kp + c * 8);
#pragma unroll
        for (int e = 0; e < 8; ++e) s += (float)q[c][e] * (float)kv[e];
    }
    return s;
}

// stage-1 two-child online softmax merge, one thread per (b,p,h)
__global__ void k_merge(const bf16* __restrict__ q, const bf16* __restrict__ k,
                        const bf16* __restrict__ v, bf16* __restrict__ out, int P) {
    int id = blockIdx.x * blockDim.x + threadIdx.x;
    if (id >= NB * P * NH) return;
    int h = id & (NH - 1);
    int row = id >> 4;                 // b*P + p
    int b = row / P, p = row - b * P;

    const bf16* qp = q + (size_t)row * DIMC + h * DHEAD;
    v8bf qv[8];
#pragma unroll
    for (int c = 0; c < 8; ++c) qv[c] = *(const v8bf*)(qp + c * 8);

    size_t kb = ((size_t)b * (2 * P) + 2 * p) * DIMC + h * DHEAD;
    float s0 = dot64(qv, k + kb) * SCALEF;
    float s1 = dot64(qv, k + kb + DIMC) * SCALEF;
    float m  = fmaxf(s0, s1);
    float e0 = __expf(s0 - m), e1 = __expf(s1 - m);
    float inv = 1.0f / (e0 + e1 + 1e-9f);
    float w0 = e0 * inv, w1 = e1 * inv;

    bf16* op = out + (size_t)row * DIMC + h * DHEAD;
#pragma unroll
    for (int c = 0; c < 8; ++c) {
        v8bf v0 = *(const v8bf*)(v + kb + c * 8);
        v8bf v1 = *(const v8bf*)(v + kb + DIMC + c * 8);
        v8bf o;
#pragma unroll
        for (int e = 0; e < 8; ++e)
            o[e] = (bf16)(w0 * (float)v0[e] + w1 * (float)v1[e]);
        *(v8bf*)(op + c * 8) = o;
    }
}

// stage-2 gather attention: self + 12 tree neighbors, one thread per (b,n,h)
__global__ void k_gather_attn(const bf16* __restrict__ Q, const bf16* __restrict__ Kall,
                              const bf16* __restrict__ Vall,
                              const int* __restrict__ idx_map,
                              const unsigned char* __restrict__ cmask,
                              bf16* __restrict__ out) {
    int id = blockIdx.x * blockDim.x + threadIdx.x;
    if (id >= NB * NSEQ * NH) return;
    int h = id & (NH - 1);
    int r = id >> 4;                   // r = b*N + n
    int b = r / NSEQ, n = r - b * NSEQ;

    const bf16* qp = Q + (size_t)r * DIMC + h * DHEAD;
    v8bf qv[8];
#pragma unroll
    for (int c = 0; c < 8; ++c) qv[c] = *(const v8bf*)(qp + c * 8);

    size_t hbase = (size_t)b * TOTN * DIMC + h * DHEAD;
    size_t rows[NLEV + 1];
    float  s[NLEV + 1];

    rows[0] = hbase + (size_t)n * DIMC;
    s[0] = dot64(qv, Kall + rows[0]) * SCALEF;
#pragma unroll
    for (int l = 0; l < NLEV; ++l) {
        int idx = idx_map[n * NLEV + l];
        idx = idx < 0 ? 0 : (idx > TOTN - 2 ? TOTN - 2 : idx);   // jnp.take clip
        rows[l + 1] = hbase + (size_t)idx * DIMC;
        bool msk = cmask[n * NLEV + l] != 0;
        s[l + 1] = msk ? -__builtin_inff()
                       : dot64(qv, Kall + rows[l + 1]) * SCALEF;
    }

    float m = s[0];
#pragma unroll
    for (int i = 1; i <= NLEV; ++i) m = fmaxf(m, s[i]);
    float wgt[NLEV + 1], den = 0.0f;
#pragma unroll
    for (int i = 0; i <= NLEV; ++i) { wgt[i] = __expf(s[i] - m); den += wgt[i]; }
    float inv = 1.0f / den;

    bf16* op = out + (size_t)r * DIMC + h * DHEAD;
#pragma unroll
    for (int c = 0; c < 8; ++c) {
        float accv[8];
#pragma unroll
        for (int e = 0; e < 8; ++e) accv[e] = 0.0f;
#pragma unroll
        for (int i = 0; i <= NLEV; ++i) {
            if (wgt[i] > 0.0f) {
                v8bf vv = *(const v8bf*)(Vall + rows[i] + c * 8);
                float wi = wgt[i] * inv;
#pragma unroll
                for (int e = 0; e < 8; ++e) accv[e] += wi * (float)vv[e];
            }
        }
        v8bf o;
#pragma unroll
        for (int e = 0; e < 8; ++e) o[e] = (bf16)accv[e];
        *(v8bf*)(op + c * 8) = o;
    }
}

// ---------------- host orchestration ----------------

extern "C" void kernel_launch(void* const* d_in, const int* in_sizes, int n_in,
                              void* d_out, int out_size, void* d_ws, size_t ws_size,
                              hipStream_t stream) {
    (void)in_sizes; (void)n_in; (void)out_size; (void)ws_size;

    const float* x = (const float*)d_in[0];
    const float* Wsrc[8] = {
        (const float*)d_in[1], (const float*)d_in[2], (const float*)d_in[3], (const float*)d_in[4],   // Wq_y Wk_y Wv_y Wo_y
        (const float*)d_in[6], (const float*)d_in[7], (const float*)d_in[8], (const float*)d_in[9] }; // Wq_x Wk_x Wv_x Wo_x
    const float* bo_y = (const float*)d_in[5];
    const float* bo_x = (const float*)d_in[10];
    const int* idx_map = (const int*)d_in[11];
    const unsigned char* cmask = (const unsigned char*)d_in[12];

    char* ws = (char*)d_ws;
    size_t off = 0;
    auto carve = [&](size_t bytes) -> void* {
        void* p = ws + off;
        off += (bytes + 255) & ~(size_t)255;
        return p;
    };
    bf16* Wt[8];
    for (int i = 0; i < 8; ++i) Wt[i] = (bf16*)carve((size_t)DIMC * DIMC * 2);
    bf16* hall = (bf16*)carve((size_t)NB * TOTN * DIMC * 2);
    bf16* pre  = (bf16*)carve((size_t)NB * (NSEQ / 2) * DIMC * 2);
    bf16* qlv  = (bf16*)carve((size_t)NB * (NSEQ / 2) * DIMC * 2);
    bf16* klv  = (bf16*)carve((size_t)NB * NSEQ * DIMC * 2);
    bf16* vlv  = (bf16*)carve((size_t)NB * NSEQ * DIMC * 2);
    bf16* mrg  = (bf16*)carve((size_t)NB * (NSEQ / 2) * DIMC * 2);
    bf16* Qx   = (bf16*)carve((size_t)NB * NSEQ * DIMC * 2);
    bf16* Kal  = (bf16*)carve((size_t)NB * TOTN * DIMC * 2);
    bf16* Val  = (bf16*)carve((size_t)NB * TOTN * DIMC * 2);
    bf16* attn = (bf16*)carve((size_t)NB * NSEQ * DIMC * 2);

    for (int i = 0; i < 8; ++i)
        k_transpose_w<<<(DIMC * DIMC + 255) / 256, 256, 0, stream>>>(Wsrc[i], Wt[i]);
    k_x_to_hall<<<(NB * NSEQ * DIMC + 255) / 256, 256, 0, stream>>>(x, hall);

    auto gemm = [&](const bf16* A, long sA, int rpb, const bf16* Wm, const float* bias,
                    void* C, long sC, int M, bool f32out) {
        dim3 g((unsigned)((M + 127) / 128), DIMC / 64);
        dim3 blk(128);
        if (f32out) {
            if (bias) k_gemm_wmma<true, true><<<g, blk, 0, stream>>>(A, sA, rpb, Wm, bias, C, sC, M);
            else      k_gemm_wmma<true, false><<<g, blk, 0, stream>>>(A, sA, rpb, Wm, bias, C, sC, M);
        } else {
            if (bias) k_gemm_wmma<false, true><<<g, blk, 0, stream>>>(A, sA, rpb, Wm, bias, C, sC, M);
            else      k_gemm_wmma<false, false><<<g, blk, 0, stream>>>(A, sA, rpb, Wm, bias, C, sC, M);
        }
    };

    // ---- stage 1: build tree, level by level, parents written into h_all concat layout
    int offCur = 0;
    for (int l = 0; l < NLEV; ++l) {
        int C = NSEQ >> l, P = C >> 1;
        int offPar = offCur + C;
        const bf16* cur = hall + (size_t)offCur * DIMC;

        k_pair_avg<<<(NB * P * (DIMC / 8) + 255) / 256, 256, 0, stream>>>(cur, pre, P);
        gemm(pre, (long)P * DIMC,    P, Wt[0], nullptr, qlv, (long)P * DIMC, NB * P, false);
        gemm(cur, (long)TOTN * DIMC, C, Wt[1], nullptr, klv, (long)C * DIMC, NB * C, false);
        gemm(cur, (long)TOTN * DIMC, C, Wt[2], nullptr, vlv, (long)C * DIMC, NB * C, false);
        k_merge<<<(NB * P * NH + 255) / 256, 256, 0, stream>>>(qlv, klv, vlv, mrg, P);
        gemm(mrg, (long)P * DIMC, P, Wt[3], bo_y,
             hall + (size_t)offPar * DIMC, (long)TOTN * DIMC, NB * P, false);
        offCur = offPar;
    }

    // ---- stage 2: projections over all nodes + gathered multi-level attention
    gemm(hall, (long)TOTN * DIMC, NSEQ, Wt[4], nullptr, Qx,  (long)NSEQ * DIMC, NB * NSEQ, false);
    gemm(hall, (long)TOTN * DIMC, TOTN, Wt[5], nullptr, Kal, (long)TOTN * DIMC, NB * TOTN, false);
    gemm(hall, (long)TOTN * DIMC, TOTN, Wt[6], nullptr, Val, (long)TOTN * DIMC, NB * TOTN, false);
    k_gather_attn<<<(NB * NSEQ * NH + 255) / 256, 256, 0, stream>>>(Qx, Kal, Val, idx_map, cmask, attn);
    gemm(attn, (long)NSEQ * DIMC, NSEQ, Wt[7], bo_x, d_out, (long)NSEQ * DIMC, NB * NSEQ, true);
}
